// WindowAttention_10857677324835
// MI455X (gfx1250) — matrix operand order repaired and checked by reading, not compiled, not run on hardware
//
#include <hip/hip_runtime.h>
#include <hip/hip_bf16.h>

typedef __attribute__((ext_vector_type(16))) __bf16 v16bf;
typedef __attribute__((ext_vector_type(8)))  __bf16 v8bf;
typedef __attribute__((ext_vector_type(8)))  float  v8f;

static constexpr int NH    = 12;      // heads
static constexpr int NTOK  = 49;      // tokens per window
static constexpr int NPAD  = 64;      // padded tokens
static constexpr int HD    = 32;      // head dim
static constexpr int CDIM  = 384;     // channels
static constexpr int BWIN  = 2048;    // windows
static constexpr int MROWS = BWIN * NTOK;   // 100352 (divisible by 128)

// ---------- helpers ----------
__device__ __forceinline__ unsigned short f2bf(float f) {
  unsigned u = __float_as_uint(f);
  u += 0x7FFFu + ((u >> 16) & 1u);   // round-to-nearest-even
  return (unsigned short)(u >> 16);
}

// low 32 bits of a generic LDS pointer == LDS byte offset (CDNA5 aperture rule)
__device__ __forceinline__ unsigned lds32(const void* p) {
  return (unsigned)(unsigned long long)p;
}

// CDNA5 async copy: global -> LDS, 16 bytes per lane, tracked by ASYNCcnt
__device__ __forceinline__ void async_ld128(unsigned lds_off, const void* gptr) {
  asm volatile("global_load_async_to_lds_b128 %0, %1, off"
               :: "v"(lds_off), "v"((unsigned long long)gptr) : "memory");
}
__device__ __forceinline__ void wait_async0() {
  asm volatile("s_wait_asynccnt 0" ::: "memory");
}

// A-fragment (16x32 bf16, MxK): lane half selects interleaved 8-element runs
__device__ __forceinline__ v16bf load_fragA(const unsigned short* base, int row,
                                            int lda, int hf, int k0) {
  const v8bf lo = *(const v8bf*)(base + (size_t)row * lda + k0 + hf * 8);
  const v8bf hi = *(const v8bf*)(base + (size_t)row * lda + k0 + 16 + hf * 8);
  return __builtin_shufflevector(lo, hi, 0,1,2,3,4,5,6,7,8,9,10,11,12,13,14,15);
}

// B-fragment (32x16 bf16, KxN) read from B^T stored row-major (N x K):
// lane = column, half selects interleaved 8-element runs
__device__ __forceinline__ v16bf load_fragB(const unsigned short* baseT, int col,
                                            int lda, int hf, int k0) {
  const v8bf lo = *(const v8bf*)(baseT + (size_t)col * lda + k0 + hf * 8);
  const v8bf hi = *(const v8bf*)(baseT + (size_t)col * lda + k0 + 16 + hf * 8);
  return __builtin_shufflevector(lo, hi, 0,1,2,3,4,5,6,7,8,9,10,11,12,13,14,15);
}

// ---------- fp32 -> bf16 conversion (vectorized) ----------
__global__ void k_f32_to_bf16(const float* __restrict__ in,
                              unsigned short* __restrict__ out, long n4) {
  long i = (long)blockIdx.x * blockDim.x + threadIdx.x;
  long stride = (long)gridDim.x * blockDim.x;
  for (; i < n4; i += stride) {
    float4 f = ((const float4*)in)[i];
    ushort4 o;
    o.x = f2bf(f.x); o.y = f2bf(f.y); o.z = f2bf(f.z); o.w = f2bf(f.w);
    ((ushort4*)out)[i] = o;
  }
}

__global__ void k_zero16(uint4* __restrict__ p, long n) {
  long i = (long)blockIdx.x * blockDim.x + threadIdx.x;
  long stride = (long)gridDim.x * blockDim.x;
  uint4 z; z.x = 0; z.y = 0; z.z = 0; z.w = 0;
  for (; i < n; i += stride) p[i] = z;
}

// ---------- QKV GEMM: (100352x384) x (384x1152)^T, scatter q/k/vT ----------
__global__ __launch_bounds__(256)
void k_qkv_gemm(const unsigned short* __restrict__ Xbf,   // M x 384
                const unsigned short* __restrict__ Wbf,   // 1152 x 384 (N x K)
                const float* __restrict__ qkv_b,          // 1152
                unsigned short* __restrict__ qbuf,        // (b,h) 64x32, q*scale
                unsigned short* __restrict__ kbuf,        // (b,h) 64x32
                unsigned short* __restrict__ vTbuf) {     // (b,h) 32x64
  __shared__ __align__(16) unsigned short As[2][128 * 40];
  __shared__ __align__(16) unsigned short Bs[2][64 * 40];
  const int tid = threadIdx.x;
  const int lane = tid & 31, wid = tid >> 5;
  const int hf = lane >> 4, r16 = lane & 15;
  const int wm = wid & 3, wn = wid >> 2;
  const int bm = blockIdx.y, bn = blockIdx.x;

  v8f acc[2][2] = {};
  const int ar = tid >> 1, ac = (tid & 1) * 16;   // A tile: 128x32
  const int br = tid >> 2, bc = (tid & 3) * 8;    // B tile: 64x32
  const unsigned short* gA = Xbf + (size_t)(bm * 128 + ar) * CDIM + ac;
  const unsigned short* gB = Wbf + (size_t)(bn * 64 + br) * CDIM + bc;
  const unsigned lA[2] = { lds32(&As[0][ar * 40 + ac]), lds32(&As[1][ar * 40 + ac]) };
  const unsigned lB[2] = { lds32(&Bs[0][br * 40 + bc]), lds32(&Bs[1][br * 40 + bc]) };

  // prologue: async DMA of K-chunk 0 into buffer 0
  async_ld128(lA[0],      gA);
  async_ld128(lA[0] + 16, gA + 8);
  async_ld128(lB[0],      gB);
  wait_async0();
  __syncthreads();

  for (int k0 = 0; k0 < CDIM; k0 += 32) {
    const int cur = (k0 >> 5) & 1, nxt = cur ^ 1;
    const unsigned short* as = &As[cur][0];
    const unsigned short* bs = &Bs[cur][0];
    if (k0 + 32 < CDIM) {
      // overlap next tile's DMA with this tile's WMMAs
      async_ld128(lA[nxt],      gA + k0 + 32);
      async_ld128(lA[nxt] + 16, gA + k0 + 40);
      async_ld128(lB[nxt],      gB + k0 + 32);
      if (k0 + 64 < CDIM) {
        __builtin_prefetch(gA + k0 + 64, 0, 3);   // warm L2 one tile further
        __builtin_prefetch(gB + k0 + 64, 0, 3);
      }
    }
    v16bf a0 = load_fragA(as, wm * 32 + r16,      40, hf, 0);
    v16bf a1 = load_fragA(as, wm * 32 + 16 + r16, 40, hf, 0);
    v16bf b0 = load_fragB(bs, wn * 32 + r16,      40, hf, 0);
    v16bf b1 = load_fragB(bs, wn * 32 + 16 + r16, 40, hf, 0);
    acc[0][0] = __builtin_amdgcn_wmma_f32_16x16x32_bf16(false, a0, false, b0, (short)0, acc[0][0], false, false);
    acc[0][1] = __builtin_amdgcn_wmma_f32_16x16x32_bf16(false, a0, false, b1, (short)0, acc[0][1], false, false);
    acc[1][0] = __builtin_amdgcn_wmma_f32_16x16x32_bf16(false, a1, false, b0, (short)0, acc[1][0], false, false);
    acc[1][1] = __builtin_amdgcn_wmma_f32_16x16x32_bf16(false, a1, false, b1, (short)0, acc[1][1], false, false);
    if (k0 + 32 < CDIM) wait_async0();
    __syncthreads();
  }

  const float scale = 0.17677669529663689f;  // 32^-0.5
  for (int mt = 0; mt < 2; ++mt)
    for (int nt = 0; nt < 2; ++nt)
      for (int v = 0; v < 8; ++v) {
        int row = bm * 128 + wm * 32 + mt * 16 + v + 8 * hf;
        int col = bn * 64 + wn * 32 + nt * 16 + r16;
        float val = acc[mt][nt][v] + qkv_b[col];
        int b   = row / NTOK;
        int tok = row - b * NTOK;
        int which = col / CDIM;                // 0=q 1=k 2=v
        int cc = col - which * CDIM;
        int h = cc >> 5, d = cc & 31;
        size_t bh = (size_t)b * NH + h;
        if (which == 0)      qbuf[bh * 2048 + tok * HD + d]  = f2bf(val * scale);
        else if (which == 1) kbuf[bh * 2048 + tok * HD + d]  = f2bf(val);
        else                 vTbuf[bh * 2048 + d * NPAD + tok] = f2bf(val);
      }
}

// ---------- attention: one block (4 waves) per (window, head) ----------
__global__ __launch_bounds__(128)
void k_attn(const unsigned short* __restrict__ qbuf,
            const unsigned short* __restrict__ kbuf,
            const unsigned short* __restrict__ vTbuf,
            const float* __restrict__ rpb,     // 169 x 12
            const float* __restrict__ mask,    // 64 x 49 x 49
            unsigned short* __restrict__ ctx)  // 100352 x 384 (bf16)
{
  __shared__ __align__(16) unsigned short Ps[NPAD * NPAD];  // softmaxed P, bf16
  const int tid = threadIdx.x, lane = tid & 31, wid = tid >> 5;
  const int hf = lane >> 4, r16 = lane & 15;
  const int bh = blockIdx.x;
  const int b = bh / NH, h = bh - b * NH;
  const unsigned short* qh = qbuf + (size_t)bh * 2048;
  const unsigned short* kh = kbuf + (size_t)bh * 2048;
  const unsigned short* vh = vTbuf + (size_t)bh * 2048;

  // S strip (16 rows x 64 cols), K = 32 in one WMMA per 16x16 tile
  v16bf aq = load_fragA(qh, wid * 16 + r16, HD, hf, 0);
  v8f s[4] = {};
  for (int nt = 0; nt < 4; ++nt) {
    v16bf bk = load_fragB(kh, nt * 16 + r16, HD, hf, 0);
    s[nt] = __builtin_amdgcn_wmma_f32_16x16x32_bf16(false, aq, false, bk, (short)0, s[nt], false, false);
  }

  // + relative position bias + window mask, pad cols -> -1e30
  const float* mrow = mask + (size_t)(b & 63) * (NTOK * NTOK);
  for (int v = 0; v < 8; ++v) {
    int r = wid * 16 + v + 8 * hf;
    int ri = r / 7, rj = r - ri * 7;
    for (int nt = 0; nt < 4; ++nt) {
      int c = nt * 16 + r16;
      if (r < NTOK && c < NTOK) {
        int ci = c / 7, cj = c - ci * 7;
        int idx = (ri - ci + 6) * 13 + (rj - cj + 6);
        s[nt][v] += rpb[idx * NH + h] + mrow[r * NTOK + c];
      } else {
        s[nt][v] = -1e30f;
      }
    }
  }

  // softmax per row: 4 regs in-lane, then butterfly across the 16-lane half
  for (int v = 0; v < 8; ++v) {
    float m = fmaxf(fmaxf(s[0][v], s[1][v]), fmaxf(s[2][v], s[3][v]));
    for (int off = 1; off <= 8; off <<= 1) m = fmaxf(m, __shfl_xor(m, off, 32));
    float e0 = __expf(s[0][v] - m), e1 = __expf(s[1][v] - m);
    float e2 = __expf(s[2][v] - m), e3 = __expf(s[3][v] - m);
    float sum = e0 + e1 + e2 + e3;
    for (int off = 1; off <= 8; off <<= 1) sum += __shfl_xor(sum, off, 32);
    float inv = 1.0f / sum;
    s[0][v] = e0 * inv; s[1][v] = e1 * inv; s[2][v] = e2 * inv; s[3][v] = e3 * inv;
  }

  // stage P (bf16) in LDS in row-major 64x64
  for (int v = 0; v < 8; ++v) {
    int row = wid * 16 + v + 8 * hf;
    for (int nt = 0; nt < 4; ++nt)
      Ps[row * NPAD + nt * 16 + r16] = f2bf(s[nt][v]);
  }
  __syncthreads();

  // O strip = P(16x64) x V(64x32): 2 K-chunks x 2 N-tiles
  v8f o[2] = {};
  for (int kc = 0; kc < 2; ++kc) {
    v16bf ap = load_fragA(Ps, wid * 16 + r16, NPAD, hf, kc * 32);
    for (int nt = 0; nt < 2; ++nt) {
      v16bf bv = load_fragB(vh, nt * 16 + r16, NPAD, hf, kc * 32);
      o[nt] = __builtin_amdgcn_wmma_f32_16x16x32_bf16(false, ap, false, bv, (short)0, o[nt], false, false);
    }
  }

  for (int v = 0; v < 8; ++v) {
    int tok = wid * 16 + v + 8 * hf;
    if (tok < NTOK)
      for (int nt = 0; nt < 2; ++nt)
        ctx[((size_t)b * NTOK + tok) * CDIM + h * HD + nt * 16 + r16] = f2bf(o[nt][v]);
  }
}

// ---------- projection GEMM: (100352x384) x (384x384)^T + bias -> f32 ----------
__global__ __launch_bounds__(256)
void k_proj_gemm(const unsigned short* __restrict__ Abf,  // M x 384 bf16
                 const unsigned short* __restrict__ Wbf,  // 384 x 384 (N x K)
                 const float* __restrict__ bias,
                 float* __restrict__ out) {
  __shared__ __align__(16) unsigned short As[2][128 * 40];
  __shared__ __align__(16) unsigned short Bs[2][64 * 40];
  const int tid = threadIdx.x;
  const int lane = tid & 31, wid = tid >> 5;
  const int hf = lane >> 4, r16 = lane & 15;
  const int wm = wid & 3, wn = wid >> 2;
  const int bm = blockIdx.y, bn = blockIdx.x;

  v8f acc[2][2] = {};
  const int ar = tid >> 1, ac = (tid & 1) * 16;
  const int br = tid >> 2, bc = (tid & 3) * 8;
  const unsigned short* gA = Abf + (size_t)(bm * 128 + ar) * CDIM + ac;
  const unsigned short* gB = Wbf + (size_t)(bn * 64 + br) * CDIM + bc;
  const unsigned lA[2] = { lds32(&As[0][ar * 40 + ac]), lds32(&As[1][ar * 40 + ac]) };
  const unsigned lB[2] = { lds32(&Bs[0][br * 40 + bc]), lds32(&Bs[1][br * 40 + bc]) };

  async_ld128(lA[0],      gA);
  async_ld128(lA[0] + 16, gA + 8);
  async_ld128(lB[0],      gB);
  wait_async0();
  __syncthreads();

  for (int k0 = 0; k0 < CDIM; k0 += 32) {
    const int cur = (k0 >> 5) & 1, nxt = cur ^ 1;
    const unsigned short* as = &As[cur][0];
    const unsigned short* bs = &Bs[cur][0];
    if (k0 + 32 < CDIM) {
      async_ld128(lA[nxt],      gA + k0 + 32);
      async_ld128(lA[nxt] + 16, gA + k0 + 40);
      async_ld128(lB[nxt],      gB + k0 + 32);
      if (k0 + 64 < CDIM) {
        __builtin_prefetch(gA + k0 + 64, 0, 3);
        __builtin_prefetch(gB + k0 + 64, 0, 3);
      }
    }
    v16bf a0 = load_fragA(as, wm * 32 + r16,      40, hf, 0);
    v16bf a1 = load_fragA(as, wm * 32 + 16 + r16, 40, hf, 0);
    v16bf b0 = load_fragB(bs, wn * 32 + r16,      40, hf, 0);
    v16bf b1 = load_fragB(bs, wn * 32 + 16 + r16, 40, hf, 0);
    acc[0][0] = __builtin_amdgcn_wmma_f32_16x16x32_bf16(false, a0, false, b0, (short)0, acc[0][0], false, false);
    acc[0][1] = __builtin_amdgcn_wmma_f32_16x16x32_bf16(false, a0, false, b1, (short)0, acc[0][1], false, false);
    acc[1][0] = __builtin_amdgcn_wmma_f32_16x16x32_bf16(false, a1, false, b0, (short)0, acc[1][0], false, false);
    acc[1][1] = __builtin_amdgcn_wmma_f32_16x16x32_bf16(false, a1, false, b1, (short)0, acc[1][1], false, false);
    if (k0 + 32 < CDIM) wait_async0();
    __syncthreads();
  }

  for (int mt = 0; mt < 2; ++mt)
    for (int nt = 0; nt < 2; ++nt)
      for (int v = 0; v < 8; ++v) {
        int row = bm * 128 + wm * 32 + mt * 16 + v + 8 * hf;
        int col = bn * 64 + wn * 32 + nt * 16 + r16;
        out[(size_t)row * CDIM + col] = acc[mt][nt][v] + bias[col];
      }
}

// ---------- launch ----------
extern "C" void kernel_launch(void* const* d_in, const int* in_sizes, int n_in,
                              void* d_out, int out_size, void* d_ws, size_t ws_size,
                              hipStream_t stream) {
  const float* x      = (const float*)d_in[0];
  const float* mask   = (const float*)d_in[1];
  const float* rpb    = (const float*)d_in[2];
  const float* qkv_w  = (const float*)d_in[3];
  const float* qkv_b  = (const float*)d_in[4];
  const float* proj_w = (const float*)d_in[5];
  const float* proj_b = (const float*)d_in[6];
  float* out = (float*)d_out;

  char* ws = (char*)d_ws;
  // byte offsets (all 16B aligned)
  unsigned short* xbf   = (unsigned short*)(ws);                 // 77,070,336 B
  unsigned short* wqkv  = (unsigned short*)(ws + 77070336);      //    884,736 B
  unsigned short* wproj = (unsigned short*)(ws + 77955072);      //    294,912 B
  unsigned short* qbuf  = (unsigned short*)(ws + 78249984);      // 100,663,296 B
  unsigned short* kbuf  = (unsigned short*)(ws + 178913280);     // 100,663,296 B
  unsigned short* vTbuf = (unsigned short*)(ws + 279576576);     // 100,663,296 B
  unsigned short* ctxbf = (unsigned short*)(ws + 380239872);     // 77,070,336 B

  // 1) fp32 -> bf16 conversions
  k_f32_to_bf16<<<2048, 256, 0, stream>>>(x,      xbf,   (long)MROWS * CDIM / 4);
  k_f32_to_bf16<<<128,  256, 0, stream>>>(qkv_w,  wqkv,  (long)(3 * CDIM) * CDIM / 4);
  k_f32_to_bf16<<<64,   256, 0, stream>>>(proj_w, wproj, (long)CDIM * CDIM / 4);

  // 2) zero q/k/vT (covers token-pad rows 49..63)
  k_zero16<<<4096, 256, 0, stream>>>((uint4*)qbuf, (long)3 * 100663296 / 16);

  // 3) QKV projection (bf16 WMMA + async DMA), scatter into per-(window,head) buffers
  dim3 gq(3 * CDIM / 64, MROWS / 128);   // (18, 784)
  k_qkv_gemm<<<gq, 256, 0, stream>>>(xbf, wqkv, qkv_b, qbuf, kbuf, vTbuf);

  // 4) windowed attention, one block per (window, head)
  k_attn<<<BWIN * NH, 128, 0, stream>>>(qbuf, kbuf, vTbuf, rpb, mask, ctxbf);

  // 5) output projection -> f32
  dim3 gp(CDIM / 64, MROWS / 128);       // (6, 784)
  k_proj_gemm<<<gp, 256, 0, stream>>>(ctxbf, wproj, proj_b, out);
}
